// TTMoEBlock_14096082666062
// MI455X (gfx1250) — compile-verified, matrix operand order checked by reading, not run on hardware
//
#include <hip/hip_runtime.h>
#include <hip/hip_bf16.h>
#include <math.h>

// Problem constants (B=2, S=2048 -> N=4096; H=1024; F=2048; E=8; top-2)
#define NTOK 4096
#define HDIM 1024
#define FDIM 2048
#define NEXP 8
#define MTILE 32           // tokens per block (2 x 16-row WMMA tiles)

typedef __attribute__((ext_vector_type(16))) __bf16 v16bf;
typedef __attribute__((ext_vector_type(8)))  float  v8f;

__device__ __forceinline__ v8f wmma_bf16(v16bf a, v16bf b, v8f c) {
  // D = A(16x32 bf16) x B(32x16 bf16) + C(16x16 f32)
  return __builtin_amdgcn_wmma_f32_16x16x32_bf16(false, a, false, b, (short)0, c, false, false);
}

// A-fragment (16x32 bf16, MxK) from LDS. lane = M%16; lane>=16 handles K+8 half.
// VGPR j holds K pair: k = kbase + (j&3)*2 + (lane>=16?8:0) + (j>>2)*16
__device__ __forceinline__ v16bf lds_a_frag(const __bf16* base, int m, int stride,
                                            int kbase, int lane) {
  const int half = (lane >> 4) & 1;
  v16bf a;
#pragma unroll
  for (int jj = 0; jj < 8; ++jj) {
    int k = kbase + ((jj & 3) << 1) + (half << 3) + ((jj >> 2) << 4);
    a[2 * jj]     = base[m * stride + k];
    a[2 * jj + 1] = base[m * stride + k + 1];
  }
  return a;
}

// B-fragment (32x16 bf16, KxN) from a global fp32 matrix stored [k][n] with
// row stride ldw. lane = N%16; same K-pair mapping as A. fp32 -> bf16 on the fly.
__device__ __forceinline__ v16bf glb_b_frag(const float* __restrict__ W, int kbase,
                                            int n, int ldw, int lane) {
  const int half = (lane >> 4) & 1;
  v16bf b;
#pragma unroll
  for (int jj = 0; jj < 8; ++jj) {
    int k = kbase + ((jj & 3) << 1) + (half << 3) + ((jj >> 2) << 4);
    const float* p = W + (size_t)k * ldw + n;
    b[2 * jj]     = (__bf16)p[0];
    b[2 * jj + 1] = (__bf16)p[ldw];
  }
  return b;
}

// ---------------------------------------------------------------------------
// Kernel 1: router. One wave32 per token: logits = x @ gate_w^T, softmax,
// top-2, renormalize. Writes router_logits (output #2) and per-token top-2.
// ---------------------------------------------------------------------------
__global__ void router_kernel(const float* __restrict__ x,
                              const float* __restrict__ gw,
                              float* __restrict__ logits_out,
                              float* __restrict__ topw,
                              int* __restrict__ topi) {
  const int lane = threadIdx.x & 31;
  const int n = blockIdx.x * (blockDim.x >> 5) + (threadIdx.x >> 5);
  if (n >= NTOK) return;
  const float* xr = x + (size_t)n * HDIM;
  float lg[NEXP];
#pragma unroll
  for (int e = 0; e < NEXP; ++e) {
    const float* g = gw + (size_t)e * HDIM;
    float s = 0.f;
    for (int h = lane; h < HDIM; h += 32) s += xr[h] * g[h];
#pragma unroll
    for (int off = 16; off > 0; off >>= 1) s += __shfl_xor(s, off, 32);
    lg[e] = s;  // every lane holds the full sum
  }
  if (lane < NEXP) logits_out[(size_t)n * NEXP + lane] = lg[lane];

  // softmax + top-2 (redundant across lanes; lane 0 writes)
  float mx = lg[0];
#pragma unroll
  for (int e = 1; e < NEXP; ++e) mx = fmaxf(mx, lg[e]);
  float p[NEXP];
#pragma unroll
  for (int e = 0; e < NEXP; ++e) p[e] = __expf(lg[e] - mx);
  int i0 = 0; float v0 = p[0];
#pragma unroll
  for (int e = 1; e < NEXP; ++e) if (p[e] > v0) { v0 = p[e]; i0 = e; }
  int i1 = -1; float v1 = -1.f;
#pragma unroll
  for (int e = 0; e < NEXP; ++e) if (e != i0 && p[e] > v1) { v1 = p[e]; i1 = e; }
  const float inv = 1.f / (v0 + v1);
  if (lane == 0) {
    topi[n * 2]     = i0;
    topi[n * 2 + 1] = i1;
    topw[n * 2]     = v0 * inv;
    topw[n * 2 + 1] = v1 * inv;
  }
}

// ---------------------------------------------------------------------------
// Kernel 2: deterministic per-expert token list build (ballot/popcount scan).
// One wave per expert. list entry = (token<<1)|rank.
// ---------------------------------------------------------------------------
__global__ void build_lists_kernel(const int* __restrict__ topi,
                                   int* __restrict__ lists,
                                   int* __restrict__ cnt) {
  const int lane = threadIdx.x & 31;
  const int e = threadIdx.x >> 5;  // 8 waves
  int base = 0;
  for (int n0 = 0; n0 < NTOK; n0 += 32) {
    const int n = n0 + lane;
    int rank = -1;
    const int a = topi[n * 2], b = topi[n * 2 + 1];
    if (a == e) rank = 0; else if (b == e) rank = 1;
    const unsigned mask = (unsigned)__ballot(rank >= 0);
    if (rank >= 0) {
      const int off = __popc(mask & ((1u << lane) - 1u));
      lists[e * NTOK + base + off] = (n << 1) | rank;
    }
    base += __popc(mask);
  }
  if (lane == 0) cnt[e] = base;
}

// ---------------------------------------------------------------------------
// Kernel 3: fused grouped GEMM + SwiGLU + down-projection. One block handles
// one expert and 32 gathered tokens (2 WMMA M-tiles -> every weight fragment
// feeds 2 WMMAs). 512 threads = 16 waves. bf16 WMMA, f32 accumulate.
// LDS: 64KB x-tile + 128KB phi-tile (~192.5KB of the 320KB WGP budget).
// ---------------------------------------------------------------------------
__global__ void __launch_bounds__(512)
moe_expert_kernel(const float* __restrict__ x,
                  const float* __restrict__ w_up,
                  const float* __restrict__ w_gate,
                  const float* __restrict__ w_down,
                  const float* __restrict__ b_up,
                  const float* __restrict__ b_gate,
                  const float* __restrict__ b_down,
                  const float* __restrict__ topw,
                  const int* __restrict__ lists,
                  const int* __restrict__ cnt,
                  float* __restrict__ contrib) {
  __shared__ __bf16 sA[MTILE * HDIM];     // 64 KB: 32 token rows of x (bf16)
  __shared__ __bf16 sPhi[MTILE * FDIM];   // 128 KB: silu(v)*u (bf16)
  __shared__ int   s_tok[MTILE];
  __shared__ int   s_rank[MTILE];
  __shared__ int   s_valid[MTILE];
  __shared__ float s_wt[MTILE];

  const int e = blockIdx.x;
  const int r = blockIdx.y;
  const int cnt_e = cnt[e];
  if (r * MTILE >= cnt_e) return;  // uniform per block

  const int tid  = threadIdx.x;
  const int lane = tid & 31;
  const int wave = tid >> 5;       // 0..15
  const int half = (lane >> 4) & 1;
  const int nl   = lane & 15;

  if (tid < MTILE) {
    const int idx = r * MTILE + tid;
    if (idx < cnt_e) {
      const int packed = lists[e * NTOK + idx];
      const int tok = packed >> 1, rk = packed & 1;
      s_tok[tid] = tok; s_rank[tid] = rk; s_valid[tid] = 1;
      s_wt[tid] = topw[tok * 2 + rk];
    } else {
      s_tok[tid] = 0; s_rank[tid] = 0; s_valid[tid] = 0; s_wt[tid] = 0.f;
    }
  }
  __syncthreads();

  // Stage gathered x rows into LDS as bf16 (coalesced).
  for (int i = tid; i < MTILE * HDIM; i += 512) {
    const int m = i >> 10, h = i & (HDIM - 1);
    sA[i] = (__bf16)x[(size_t)s_tok[m] * HDIM + h];
  }
  __syncthreads();

  // ---- Phase B: u = x@w_up, v = x@w_gate; phi = silu(v+bg)*(u+bu) ----------
  // 128 F-tiles over 16 waves -> 8 per wave; 2 per pass, 4 passes.
  const float* Wu = w_up   + (size_t)e * HDIM * FDIM;  // [H][F]
  const float* Wg = w_gate + (size_t)e * HDIM * FDIM;
  for (int pass = 0; pass < 4; ++pass) {
    const int ft0 = wave * 8 + pass * 2;
    v8f accU[2][2], accV[2][2];  // [mtile][ctile]
#pragma unroll
    for (int mt = 0; mt < 2; ++mt)
#pragma unroll
      for (int j = 0; j < 2; ++j) { accU[mt][j] = (v8f){}; accV[mt][j] = (v8f){}; }
    for (int kt = 0; kt < HDIM / 32; ++kt) {
      const v16bf a0 = lds_a_frag(sA, nl,      HDIM, kt * 32, lane);
      const v16bf a1 = lds_a_frag(sA, nl + 16, HDIM, kt * 32, lane);
#pragma unroll
      for (int j = 0; j < 2; ++j) {
        const int f = (ft0 + j) * 16 + nl;
        const v16bf bu = glb_b_frag(Wu, kt * 32, f, FDIM, lane);
        accU[0][j] = wmma_bf16(a0, bu, accU[0][j]);
        accU[1][j] = wmma_bf16(a1, bu, accU[1][j]);
        const v16bf bg = glb_b_frag(Wg, kt * 32, f, FDIM, lane);
        accV[0][j] = wmma_bf16(a0, bg, accV[0][j]);
        accV[1][j] = wmma_bf16(a1, bg, accV[1][j]);
      }
    }
#pragma unroll
    for (int j = 0; j < 2; ++j) {
      const int f = (ft0 + j) * 16 + nl;        // fixed N column per lane
      const float bu = b_up[e * FDIM + f];
      const float bg = b_gate[e * FDIM + f];
#pragma unroll
      for (int mt = 0; mt < 2; ++mt) {
#pragma unroll
        for (int i = 0; i < 8; ++i) {
          const float u = accU[mt][j][i] + bu;
          const float v = accV[mt][j][i] + bg;
          const float s = v / (1.f + __expf(-v));     // silu
          const int m = mt * 16 + i + half * 8;       // M row of this element
          sPhi[m * FDIM + f] = (__bf16)(s * u);
        }
      }
    }
  }
  __syncthreads();

  // ---- Phase C: y = phi @ w_down + b_down; scale by combine weight --------
  // 64 H-tiles over 16 waves -> 4 per wave.
  const float* Wd = w_down + (size_t)e * FDIM * HDIM;  // [F][H]
  const int ht0 = wave * 4;
  v8f acc[2][4];
#pragma unroll
  for (int j = 0; j < 4; ++j) {
    const float bd = b_down[e * HDIM + (ht0 + j) * 16 + nl];
    v8f t;
#pragma unroll
    for (int i = 0; i < 8; ++i) t[i] = bd;
    acc[0][j] = t; acc[1][j] = t;
  }
  for (int kt = 0; kt < FDIM / 32; ++kt) {
    const v16bf a0 = lds_a_frag(sPhi, nl,      FDIM, kt * 32, lane);
    const v16bf a1 = lds_a_frag(sPhi, nl + 16, FDIM, kt * 32, lane);
#pragma unroll
    for (int j = 0; j < 4; ++j) {
      const int h = (ht0 + j) * 16 + nl;
      const v16bf b = glb_b_frag(Wd, kt * 32, h, HDIM, lane);
      acc[0][j] = wmma_bf16(a0, b, acc[0][j]);
      acc[1][j] = wmma_bf16(a1, b, acc[1][j]);
    }
  }
  // Each (token,rank) pair is owned by exactly one block row -> plain stores.
#pragma unroll
  for (int j = 0; j < 4; ++j) {
    const int h = (ht0 + j) * 16 + nl;
#pragma unroll
    for (int mt = 0; mt < 2; ++mt) {
#pragma unroll
      for (int i = 0; i < 8; ++i) {
        const int m = mt * 16 + i + half * 8;
        if (s_valid[m]) {
          const float val = acc[mt][j][i] * s_wt[m];
          contrib[(size_t)s_rank[m] * NTOK * HDIM + (size_t)s_tok[m] * HDIM + h] = val;
        }
      }
    }
  }
}

// ---------------------------------------------------------------------------
// Kernel 4: deterministic combine of the two expert contributions per token.
// ---------------------------------------------------------------------------
__global__ void combine_kernel(const float* __restrict__ contrib,
                               float* __restrict__ out) {
  const size_t total = (size_t)NTOK * HDIM;
  for (size_t i = (size_t)blockIdx.x * blockDim.x + threadIdx.x; i < total;
       i += (size_t)gridDim.x * blockDim.x) {
    out[i] = contrib[i] + contrib[total + i];
  }
}

// ---------------------------------------------------------------------------
extern "C" void kernel_launch(void* const* d_in, const int* in_sizes, int n_in,
                              void* d_out, int out_size, void* d_ws, size_t ws_size,
                              hipStream_t stream) {
  const float* x      = (const float*)d_in[0];  // [B,S,H]
  const float* gate_w = (const float*)d_in[1];  // [E,H]
  const float* w_up   = (const float*)d_in[2];  // [E,H,F]
  const float* w_gate = (const float*)d_in[3];  // [E,H,F]
  const float* w_down = (const float*)d_in[4];  // [E,F,H]
  const float* b_up   = (const float*)d_in[5];  // [E,F]
  const float* b_gate = (const float*)d_in[6];  // [E,F]
  const float* b_down = (const float*)d_in[7];  // [E,H]

  float* out        = (float*)d_out;                   // [N,H]
  float* logits_out = out + (size_t)NTOK * HDIM;       // [N,E]

  // Workspace layout
  char* ws = (char*)d_ws;
  float* topw    = (float*)(ws);                 // N*2 f32        (32 KB)
  int*   topi    = (int*)(ws + 32768);           // N*2 i32        (32 KB)
  int*   cntp    = (int*)(ws + 65536);           // E i32
  int*   lists   = (int*)(ws + 65792);           // E*N i32        (128 KB)
  float* contrib = (float*)(ws + 196864);        // 2*N*H f32      (32 MB)

  router_kernel<<<NTOK / 8, 256, 0, stream>>>(x, gate_w, logits_out, topw, topi);
  build_lists_kernel<<<1, 256, 0, stream>>>(topi, lists, cntp);
  dim3 g3(NEXP, NTOK / MTILE);
  moe_expert_kernel<<<g3, 512, 0, stream>>>(x, w_up, w_gate, w_down,
                                            b_up, b_gate, b_down,
                                            topw, lists, cntp, contrib);
  combine_kernel<<<4096, 256, 0, stream>>>(contrib, out);
}
